// ILMSPAM_Module_38233798869158
// MI455X (gfx1250) — matrix operand order repaired and checked by reading, not compiled, not run on hardware
//
#include <hip/hip_runtime.h>
#include <hip/hip_bf16.h>

// CDNA5 wave32 WMMA types
typedef __attribute__((ext_vector_type(16))) _Float16 v16h;
typedef __attribute__((ext_vector_type(8)))  _Float16 v8h;
typedef __attribute__((ext_vector_type(4)))  _Float16 v4h;
typedef __attribute__((ext_vector_type(8)))  float    v8f;

// ---------------------------------------------------------------------------
static __device__ __forceinline__ float delu_f(float x) {
  float r  = x > 0.0f ? x : 0.0f;
  float mn = x < 0.0f ? x : 0.0f;
  return 10.0f * r + __expf(10.0f * mn);
}

static __device__ __forceinline__ v4h cvt4(float4 f) {
  v4h h;
  h[0] = (_Float16)f.x; h[1] = (_Float16)f.y;
  h[2] = (_Float16)f.z; h[3] = (_Float16)f.w;
  return h;
}

// ---------------------------------------------------------------------------
// Unified LDS tile layout: [out-dim(64)][K] with ld multiple of 8 halves, so
// every lane's fragment is contiguous 16B runs -> ds_load_b128.
// A (16x32 MxK): lane L row M=L&15; halves [0..7] = K=kb..kb+7 (kb=8 if L>=16),
//                halves [8..15] = K=16+kb..16+kb+7.
// B (32x16 KxN) staged as [n][k]: lane L col N=L&15; same K order as A:
//                halves [0..7] = K=kb..kb+7, halves [8..15] = K=16+kb..16+kb+7.
// C/D (16x16 f32): VGPR r, lane L: N=L&15, M = r + 8*(L>=16).
// ---------------------------------------------------------------------------
static __device__ __forceinline__ v16h frag_A(const _Float16* s, int ld, int ro,
                                              int kc, int lane) {
  const _Float16* r = s + (ro + (lane & 15)) * ld + kc + ((lane & 16) ? 8 : 0);
  v8h lo = *(const v8h*)(r);
  v8h hi = *(const v8h*)(r + 16);
  return __builtin_shufflevector(lo, hi, 0, 1, 2, 3, 4, 5, 6, 7,
                                 8, 9, 10, 11, 12, 13, 14, 15);
}

static __device__ __forceinline__ v16h frag_B(const _Float16* s, int ld, int no,
                                              int kc, int lane) {
  const _Float16* r = s + (no + (lane & 15)) * ld + kc + ((lane & 16) ? 8 : 0);
  v8h lo = *(const v8h*)(r);
  v8h hi = *(const v8h*)(r + 16);
  return __builtin_shufflevector(lo, hi, 0, 1, 2, 3, 4, 5, 6, 7,
                                 8, 9, 10, 11, 12, 13, 14, 15);
}

static __device__ __forceinline__ v8f wmma16(v16h a, v16h b, v8f c) {
  return __builtin_amdgcn_wmma_f32_16x16x32_f16(false, a, false, b, (short)0, c,
                                                false, false);
}

#define LD 72   // LDS tile row pitch in halves (144B, 16B-aligned rows)

// ===========================================================================
// Kernel 0: transpose+convert  x f32 [b][c][n] -> xT f16 [b][n][c]
// ===========================================================================
__global__ __launch_bounds__(256) void xpose_kernel(const float* __restrict__ x,
                                                    _Float16* __restrict__ xT) {
  const int C = 512, N = 4096;
  __shared__ _Float16 t[64 * LD];           // [c][n]
  const int n0 = blockIdx.x * 64, c0 = blockIdx.y * 64, b = blockIdx.z;
  const float* xb = x + (size_t)b * C * N;
  const int tid = threadIdx.x;
#pragma unroll
  for (int e = 0; e < 4; ++e) {             // 1024 float4 reads, coalesced on n
    int q = tid + 256 * e;
    int row = q >> 4, col = (q & 15) * 4;
    float4 f = *(const float4*)(xb + (size_t)(c0 + row) * N + n0 + col);
    *(v4h*)(t + row * LD + col) = cvt4(f);
  }
  __syncthreads();
  _Float16* xTb = xT + (size_t)b * N * 512;
#pragma unroll
  for (int e = 0; e < 2; ++e) {             // 512 v8h writes, coalesced on c
    int q = tid + 256 * e;
    int n = q >> 3, seg = (q & 7) * 8;
    v8h v;
#pragma unroll
    for (int j = 0; j < 8; ++j) v[j] = t[(seg + j) * LD + n];
    *(v8h*)(xTb + (size_t)(n0 + n) * 512 + c0 + seg) = v;
  }
}

// ===========================================================================
// Kernel 1: fused QKV projection.  [640,512] x [512,4096] per batch.
// M0==0 -> Q (delu) -> QfT [n][64];  M0==64 -> K (delu) -> Kf [m][n];
// M0>=128 -> V -> Vw [c][n].  Block tile 64x64, 8 waves 4x2, K-stage 64.
// ===========================================================================
__global__ __launch_bounds__(256) void qkv_gemm(
    const _Float16* __restrict__ xT,
    const float* __restrict__ wq, const float* __restrict__ bq,
    const float* __restrict__ wk, const float* __restrict__ bk,
    const float* __restrict__ wv, const float* __restrict__ bv,
    _Float16* __restrict__ QfT, _Float16* __restrict__ Kf,
    _Float16* __restrict__ Vw)
{
  const int C = 512, N = 4096;
  __shared__ _Float16 sA[64 * LD];          // [m][k]
  __shared__ _Float16 sB[64 * LD];          // [n][k]
  const int n0 = blockIdx.x * 64, M0 = blockIdx.y * 64, b = blockIdx.z;
  // Block-uniform weight base: kills per-element divergence.
  const float* wbase = (M0 == 0) ? wq : (M0 == 64) ? wk
                                      : (wv + (size_t)(M0 - 128) * 512);
  const _Float16* xTb = xT + (size_t)b * N * 512;
  const int tid = threadIdx.x, lane = tid & 31, w = tid >> 5;
  const int mw = (w >> 1) * 16, nw = (w & 1) * 32;
  v8f acc0 = {}; v8f acc1 = {};

  for (int kb = 0; kb < C; kb += 64) {
#pragma unroll
    for (int e = 0; e < 4; ++e) {           // A: 64x64 f32 weights -> f16
      int q = tid + 256 * e;
      int row = q >> 4, col = (q & 15) * 4;
      float4 f = *(const float4*)(wbase + (size_t)row * 512 + kb + col);
      *(v4h*)(sA + row * LD + col) = cvt4(f);
    }
#pragma unroll
    for (int e = 0; e < 2; ++e) {           // B: straight b128 row copies of xT
      int q = tid + 256 * e;
      int row = q >> 3, seg = (q & 7) * 8;
      *(v8h*)(sB + row * LD + seg) =
          *(const v8h*)(xTb + (size_t)(n0 + row) * 512 + kb + seg);
    }
    __syncthreads();
#pragma unroll
    for (int kc = 0; kc < 64; kc += 32) {
      v16h a  = frag_A(sA, LD, mw, kc, lane);
      v16h b0 = frag_B(sB, LD, nw, kc, lane);
      v16h b1 = frag_B(sB, LD, nw + 16, kc, lane);
      acc0 = wmma16(a, b0, acc0);
      acc1 = wmma16(a, b1, acc1);
    }
    __syncthreads();
  }

  const int mloc = mw + ((lane & 16) ? 8 : 0);
  const int ncol = lane & 15;
  if (M0 == 0) {                            // Q path: vector 16B stores to QfT
    _Float16* Qb = QfT + (size_t)b * N * 64;
#pragma unroll
    for (int t = 0; t < 2; ++t) {
      int gn = n0 + nw + t * 16 + ncol;
      v8h vv;
#pragma unroll
      for (int r = 0; r < 8; ++r)
        vv[r] = (_Float16)delu_f((t ? acc1[r] : acc0[r]) + bq[mloc + r]);
      *(v8h*)(Qb + (size_t)gn * 64 + mloc) = vv;
    }
  } else if (M0 == 64) {                    // K path
    _Float16* Kb = Kf + (size_t)b * 64 * N;
#pragma unroll
    for (int r = 0; r < 8; ++r) {
      int m = mloc + r;
      float bias = bk[m];
#pragma unroll
      for (int t = 0; t < 2; ++t) {
        int gn = n0 + nw + t * 16 + ncol;
        Kb[(size_t)m * N + gn] = (_Float16)delu_f((t ? acc1[r] : acc0[r]) + bias);
      }
    }
  } else {                                  // V path
    _Float16* Vb = Vw + (size_t)b * 512 * N;
    const int cb = M0 - 128;
#pragma unroll
    for (int r = 0; r < 8; ++r) {
      int c = cb + mloc + r;
      float bias = bv[c];
#pragma unroll
      for (int t = 0; t < 2; ++t) {
        int gn = n0 + nw + t * 16 + ncol;
        Vb[(size_t)c * N + gn] = (_Float16)((t ? acc1[r] : acc0[r]) + bias);
      }
    }
  }
}

// ===========================================================================
// Kernel 2: KV partials.  KV[m,c] = sum_n Kf[m,n]*V[c,n].  Split-K over n
// (8 splits of 512).  Both LDS tiles are straight b128 row copies.
// ===========================================================================
__global__ __launch_bounds__(256) void kv_gemm(
    const _Float16* __restrict__ Kf, const _Float16* __restrict__ Vw,
    float* __restrict__ KVp)                // [8][B][64][512]
{
  const int N = 4096;
  __shared__ _Float16 sA[64 * LD];          // [m][n]
  __shared__ _Float16 sB[64 * LD];          // [c][n]
  const int c0 = blockIdx.x * 64, sp = blockIdx.y, b = blockIdx.z;
  const int tid = threadIdx.x, lane = tid & 31, w = tid >> 5;
  const int mw = (w >> 1) * 16, nw = (w & 1) * 32;
  const _Float16* Kb = Kf + (size_t)b * 64 * N;
  const _Float16* Vb = Vw + (size_t)b * 512 * N;
  const int nbeg = sp * 512, nend = nbeg + 512;
  v8f acc0 = {}; v8f acc1 = {};

  for (int n0 = nbeg; n0 < nend; n0 += 64) {
#pragma unroll
    for (int e = 0; e < 2; ++e) {
      int q = tid + 256 * e;
      int row = q >> 3, seg = (q & 7) * 8;
      *(v8h*)(sA + row * LD + seg) =
          *(const v8h*)(Kb + (size_t)row * N + n0 + seg);
      *(v8h*)(sB + row * LD + seg) =
          *(const v8h*)(Vb + (size_t)(c0 + row) * N + n0 + seg);
      if (e == 0 && n0 + 64 < nend)
        __builtin_prefetch(Vb + (size_t)(c0 + row) * N + n0 + 64 + seg, 0, 1);
    }
    __syncthreads();
#pragma unroll
    for (int kc = 0; kc < 64; kc += 32) {
      v16h a  = frag_A(sA, LD, mw, kc, lane);
      v16h b0 = frag_B(sB, LD, nw, kc, lane);
      v16h b1 = frag_B(sB, LD, nw + 16, kc, lane);
      acc0 = wmma16(a, b0, acc0);
      acc1 = wmma16(a, b1, acc1);
    }
    __syncthreads();
  }

  float* out = KVp + ((size_t)sp * 8 + b) * 64 * 512;
  const int mloc = mw + ((lane & 16) ? 8 : 0);
  const int ncol = lane & 15;
#pragma unroll
  for (int r = 0; r < 8; ++r)
#pragma unroll
    for (int t = 0; t < 2; ++t)
      out[(mloc + r) * 512 + (c0 + nw + t * 16 + ncol)] = t ? acc1[r] : acc0[r];
}

__global__ __launch_bounds__(256) void kv_reduce(const float* __restrict__ KVp,
                                                 float* __restrict__ KV) {
  const size_t stride = (size_t)8 * 64 * 512;
  int i = blockIdx.x * 256 + threadIdx.x;
  float s = 0.f;
#pragma unroll
  for (int p = 0; p < 8; ++p) s += KVp[(size_t)p * stride + i];
  KV[i] = s;
}

// ===========================================================================
// Ksum[b,m] = sum_n Kf[b,m,n]   (v8h vector loads)
// ===========================================================================
__global__ __launch_bounds__(256) void ksum_kernel(const _Float16* __restrict__ Kf,
                                                   float* __restrict__ Ksum) {
  const int N = 4096;
  __shared__ float red[256];
  const int m = blockIdx.x, b = blockIdx.y;
  const _Float16* row = Kf + ((size_t)b * 64 + m) * N;
  float s = 0.f;
#pragma unroll
  for (int e = 0; e < 2; ++e) {
    v8h v = *(const v8h*)(row + (threadIdx.x + 256 * e) * 8);
#pragma unroll
    for (int j = 0; j < 8; ++j) s += (float)v[j];
  }
  red[threadIdx.x] = s;
  __syncthreads();
  for (int off = 128; off > 0; off >>= 1) {
    if (threadIdx.x < off) red[threadIdx.x] += red[threadIdx.x + off];
    __syncthreads();
  }
  if (threadIdx.x == 0) Ksum[b * 64 + m] = red[0];
}

// ===========================================================================
// norm[b,n] = 1 / sum_m QfT[b,n,m]*(Ksum[b,m]+EPS)  (contiguous v8h loads)
// ===========================================================================
__global__ __launch_bounds__(256) void norm_kernel(const _Float16* __restrict__ QfT,
                                                   const float* __restrict__ Ksum,
                                                   float* __restrict__ nrm) {
  const int N = 4096;
  __shared__ float ks[64];
  const int b = blockIdx.y;
  if (threadIdx.x < 64) ks[threadIdx.x] = Ksum[b * 64 + threadIdx.x] + 1e-10f;
  __syncthreads();
  const int n = blockIdx.x * 256 + threadIdx.x;
  const _Float16* q = QfT + ((size_t)b * N + n) * 64;
  float s = 0.f;
#pragma unroll
  for (int e = 0; e < 8; ++e) {
    v8h v = *(const v8h*)(q + e * 8);
#pragma unroll
    for (int j = 0; j < 8; ++j) s += (float)v[j] * ks[e * 8 + j];
  }
  nrm[(size_t)b * N + n] = 1.0f / s;
}

// ===========================================================================
// Kernel 4: out[b,c,n] = x + gamma*norm[n]*sum_m KV[m,c]*QfT[n,m]
// Single K-stage (K=64, 4 WMMAs), fused residual epilogue.
// ===========================================================================
__global__ __launch_bounds__(256) void attn_gemm(
    const float* __restrict__ x, const _Float16* __restrict__ QfT,
    const float* __restrict__ KV, const float* __restrict__ nrm,
    const float* __restrict__ gamma, float* __restrict__ out)
{
  const int N = 4096, C = 512;
  __shared__ _Float16 sA[64 * LD];          // [c_local][m]
  __shared__ _Float16 sB[64 * LD];          // [n][m]
  const int n0 = blockIdx.x * 64, c0 = blockIdx.y * 64, b = blockIdx.z;
  const int tid = threadIdx.x, lane = tid & 31, w = tid >> 5;
  const int mw = (w >> 1) * 16, nw = (w & 1) * 32;
  const float* KVb = KV + (size_t)b * 64 * 512;
  const _Float16* Qb = QfT + (size_t)b * N * 64;
  v8f acc0 = {}; v8f acc1 = {};

#pragma unroll
  for (int e = 0; e < 16; ++e) {            // A: KV^T (f32->f16), coalesced on c
    int idx = tid + 256 * e;
    int cl = idx & 63, m = idx >> 6;
    sA[cl * LD + m] = (_Float16)KVb[(size_t)m * 512 + c0 + cl];
  }
#pragma unroll
  for (int e = 0; e < 2; ++e) {             // B: straight b128 row copies of QfT
    int q = tid + 256 * e;
    int row = q >> 3, seg = (q & 7) * 8;
    *(v8h*)(sB + row * LD + seg) =
        *(const v8h*)(Qb + (size_t)(n0 + row) * 64 + seg);
  }
  __syncthreads();
#pragma unroll
  for (int kc = 0; kc < 64; kc += 32) {
    v16h a  = frag_A(sA, LD, mw, kc, lane);
    v16h b0 = frag_B(sB, LD, nw, kc, lane);
    v16h b1 = frag_B(sB, LD, nw + 16, kc, lane);
    acc0 = wmma16(a, b0, acc0);
    acc1 = wmma16(a, b1, acc1);
  }

  const float g = gamma[0];
  const int mloc = mw + ((lane & 16) ? 8 : 0);
  const int ncol = lane & 15;
#pragma unroll
  for (int t = 0; t < 2; ++t) {
    int n = n0 + nw + t * 16 + ncol;
    float sc = g * nrm[(size_t)b * N + n];
#pragma unroll
    for (int r = 0; r < 8; ++r) {
      int c = c0 + mloc + r;
      size_t off = ((size_t)b * C + c) * N + n;
      float a = t ? acc1[r] : acc0[r];
      out[off] = x[off] + sc * a;
    }
  }
}

// ===========================================================================
extern "C" void kernel_launch(void* const* d_in, const int* in_sizes, int n_in,
                              void* d_out, int out_size, void* d_ws, size_t ws_size,
                              hipStream_t stream) {
  const float* x     = (const float*)d_in[0];
  const float* wq    = (const float*)d_in[1];
  const float* bq    = (const float*)d_in[2];
  const float* wk    = (const float*)d_in[3];
  const float* bk    = (const float*)d_in[4];
  const float* wv    = (const float*)d_in[5];
  const float* bv    = (const float*)d_in[6];
  const float* gamma = (const float*)d_in[7];
  float* out = (float*)d_out;

  const int B = 8, C = 512, N = 4096, D = 64;

  // Workspace carve-up (~81.3 MB), all offsets 16B-aligned by construction.
  char* p = (char*)d_ws;
  _Float16* xT  = (_Float16*)p; p += (size_t)B * N * C * sizeof(_Float16); // 32 MB
  _Float16* QfT = (_Float16*)p; p += (size_t)B * N * D * sizeof(_Float16); // 4 MB
  _Float16* Kf  = (_Float16*)p; p += (size_t)B * D * N * sizeof(_Float16); // 4 MB
  _Float16* Vw  = (_Float16*)p; p += (size_t)B * C * N * sizeof(_Float16); // 32 MB
  float* KVp  = (float*)p; p += (size_t)8 * B * D * C * sizeof(float);     // 8 MB
  float* KV   = (float*)p; p += (size_t)B * D * C * sizeof(float);         // 1 MB
  float* Ksum = (float*)p; p += (size_t)B * D * sizeof(float);             // 2 KB
  float* nrm  = (float*)p; p += (size_t)B * N * sizeof(float);             // 128 KB

  xpose_kernel<<<dim3(N / 64, C / 64, B), 256, 0, stream>>>(x, xT);
  qkv_gemm <<<dim3(N / 64, 640 / 64, B), 256, 0, stream>>>(xT, wq, bq, wk, bk,
                                                           wv, bv, QfT, Kf, Vw);
  kv_gemm  <<<dim3(C / 64, 8, B),       256, 0, stream>>>(Kf, Vw, KVp);
  kv_reduce<<<dim3((B * D * C) / 256),  256, 0, stream>>>(KVp, KV);
  ksum_kernel<<<dim3(D, B),             256, 0, stream>>>(Kf, Ksum);
  norm_kernel<<<dim3(N / 256, B),       256, 0, stream>>>(QfT, Ksum, nrm);
  attn_gemm<<<dim3(N / 64, C / 64, B),  256, 0, stream>>>(x, QfT, KV, nrm,
                                                          gamma, out);
}
